// CityNamePredictor_58841051955531
// MI455X (gfx1250) — matrix operand order repaired and checked by reading, not compiled, not run on hardware
//
#include <hip/hip_runtime.h>
#include <cstdint>
#include <cstddef>

// ---------------------------------------------------------------------------
// Vanilla tanh-RNN: xin = x@W_ih^T + b_ih (big GEMM), 512-step recurrence
// h = tanh(xin_t + h@W_hh^T + b_hh), then sigmoid(h_T @ W_lin^T + b_lin).
// All GEMMs via V_WMMA_F32_16X16X32_F16 (f16 in, f32 accumulate).
// W_hh / W_ih / W_lin B-fragments live in VGPRs; A-fragments are preloaded
// 8-deep so the dependent WMMA chain never drains the LDS pipe.
// ---------------------------------------------------------------------------

#define T_STEPS 512
#define BATCH   1024
#define IN_DIM  256
#define HID     256
#define OUT_DIM 1024
#define PROJ_MT 4            // M-tiles per projection block

typedef __attribute__((ext_vector_type(16))) _Float16 v16h;
typedef __attribute__((ext_vector_type(8)))  float    v8f;

union HFrag { v16h v; uint4 q[2]; };
union CFrag { v8f  v; float f[8]; };
union O8    { _Float16 h[8]; uint4 q; };   // one lane's 8 C values, 16 bytes

// A fragment, 16x32 f16, A row-major [M][ld] in halves.
// Lanes 0-15: row M=lane, K = k0+0..7 and k0+16..23.
// Lanes 16-31: row M=lane-16, K = k0+8..15 and k0+24..31.
__device__ __forceinline__ v16h load_a_frag(const _Float16* base, int ld,
                                            int lane, int k0) {
  const int m  = lane & 15;
  const int hs = lane >> 4;
  HFrag f;
  const _Float16* p = base + m * ld + k0 + hs * 8;
  f.q[0] = *reinterpret_cast<const uint4*>(p);
  f.q[1] = *reinterpret_cast<const uint4*>(p + 16);
  return f.v;
}

// B fragment, 32x16 f16, taken from row-major weight W[N][ld]: B[k][n]=W[n][k].
// Lane (n = lane&15) reads W row n at K = k0+0..7 and k0+16..23 (lanes 0-15)
// or K = k0+8..15 and k0+24..31 (lanes 16-31).
__device__ __forceinline__ v16h load_b_frag(const _Float16* base, int ld,
                                            int lane, int k0) {
  const int n    = lane & 15;
  const int koff = k0 + ((lane >> 4) << 3);
  HFrag f;
  const _Float16* p = base + n * ld + koff;
  f.q[0] = *reinterpret_cast<const uint4*>(p);
  f.q[1] = *reinterpret_cast<const uint4*>(p + 16);
  return f.v;
}

// ---------------------------------------------------------------------------
// fp32 -> f16 conversion (weights only; x converted on the fly in proj).
// ---------------------------------------------------------------------------
__global__ void cvt_f32_f16_kernel(const float* __restrict__ s,
                                   _Float16* __restrict__ d, int n) {
  int i = blockIdx.x * blockDim.x + threadIdx.x;
  if (i < n) d[i] = (_Float16)s[i];
}

// ---------------------------------------------------------------------------
// Kernel A: xin = x @ W_ih^T + b_ih, stored f16 in C-FRAGMENT-SWIZZLED order:
// per 16-row M tile: [wave(16)][lane(32)][r(8)] halves (8 KB per tile). In
// this layout each wave's result is 512 B contiguous, so the output is a
// direct coalesced global_store_b128 from registers (no LDS round-trip).
// Block = 4 M-tiles; W_ih B-fragments hoisted into VGPRs; 16 waves = 16 N-tiles.
// ---------------------------------------------------------------------------
__global__ __launch_bounds__(512) void rnn_proj_kernel(
    const float* __restrict__ x, const _Float16* __restrict__ wih,
    const float* __restrict__ b_ih, _Float16* __restrict__ xin) {
  __shared__ _Float16 sx[16 * IN_DIM];    // 8 KB: x tile as f16

  const int tid   = threadIdx.x;
  const int lane  = tid & 31;
  const int wave  = tid >> 5;
  const int n     = lane & 15;
  const int nbase = wave * 16;

  // hoist all 8 B fragments of this wave's W_ih N-tile (64 VGPRs)
  HFrag bf[8];
#pragma unroll
  for (int k = 0; k < 8; ++k)
    bf[k].v = load_b_frag(wih + nbase * IN_DIM, IN_DIM, lane, k * 32);
  const float bias = b_ih[nbase + n];

  for (int mt = 0; mt < PROJ_MT; ++mt) {
    const size_t mbase = ((size_t)blockIdx.x * PROJ_MT + mt) * 16;

    {  // stage + convert 16x256 f32 -> f16 (8 contiguous elems per thread)
      const float* src = x + mbase * IN_DIM;
      const int    b0  = tid * 8;
#pragma unroll
      for (int e = 0; e < 8; ++e) sx[b0 + e] = (_Float16)src[b0 + e];
    }
    __syncthreads();

    // preload all 8 A fragments, then run the WMMA chain
    HFrag af[8];
#pragma unroll
    for (int k = 0; k < 8; ++k) af[k].v = load_a_frag(sx, IN_DIM, lane, k * 32);

    CFrag c;
#pragma unroll
    for (int r = 0; r < 8; ++r) c.f[r] = bias;
#pragma unroll
    for (int k = 0; k < 8; ++k)
      c.v = __builtin_amdgcn_wmma_f32_16x16x32_f16(false, af[k].v, false,
                                                   bf[k].v, (short)0, c.v,
                                                   false, false);

    // direct coalesced store of this wave's swizzled 512 B slice
    O8 o;
#pragma unroll
    for (int r = 0; r < 8; ++r) o.h[r] = (_Float16)c.f[r];
    *reinterpret_cast<uint4*>(xin + mbase * HID + wave * 256 + lane * 8) = o.q;

    __syncthreads();   // sx reads done before next tile overwrites it
  }
}

// ---------------------------------------------------------------------------
// Kernel B: the sequential recurrence. One block owns 16 batch rows for all
// 512 timesteps. W_hh B-fragments live in VGPRs (time-invariant); LDS holds
// only double-buffered h (2x8 KB) and double-buffered xin tiles (2x8 KB).
// xin tile for step t+1 is fetched to registers while step t computes.
// ---------------------------------------------------------------------------
__global__ __launch_bounds__(512) void rnn_recur_kernel(
    const _Float16* __restrict__ xin, const _Float16* __restrict__ whh,
    const float* __restrict__ b_hh, _Float16* __restrict__ hT) {
  __shared__ _Float16 sh[2][16 * HID];    // h double buffer, row-major
  __shared__ _Float16 sxin[2][16 * HID];  // xin tile double buffer, swizzled

  const int tid   = threadIdx.x;
  const int lane  = tid & 31;
  const int wave  = tid >> 5;
  const int n     = lane & 15;
  const int hs    = lane >> 4;
  const int nbase = wave * 16;
  const int mbase = blockIdx.x * 16;
  const float bias = b_hh[nbase + n];

  // hoist all 8 B fragments of this wave's W_hh N-tile (64 VGPRs)
  HFrag bf[8];
#pragma unroll
  for (int k = 0; k < 8; ++k)
    bf[k].v = load_b_frag(whh + nbase * HID, HID, lane, k * 32);

  {  // h0 = 0 and prologue load of xin tile for t=0
    uint4 z; z.x = z.y = z.z = z.w = 0u;
    reinterpret_cast<uint4*>(sh[0])[tid] = z;
    reinterpret_cast<uint4*>(sxin[0])[tid] =
        reinterpret_cast<const uint4*>(xin + (size_t)mbase * HID)[tid];
  }
  __syncthreads();

  int cur = 0;
  for (int t = 0; t < T_STEPS; ++t) {
    const int nxt = cur ^ 1;

    // issue next tile's global load early (overlaps with WMMAs below)
    uint4 xreg;
    if (t + 1 < T_STEPS) {
      xreg = reinterpret_cast<const uint4*>(
          xin + ((size_t)(t + 1) * BATCH + mbase) * HID)[tid];
    } else {
      xreg.x = xreg.y = xreg.z = xreg.w = 0u;
    }
    // warm L2 for t+2 (emits global_prefetch_b8)
    if (wave == 0 && t + 2 < T_STEPS) {
      const char* pf = reinterpret_cast<const char*>(
          xin + ((size_t)(t + 2) * BATCH + mbase) * HID);
      __builtin_prefetch(pf + lane * 256, 0, 1);
    }

    // c = xin_t (one 16 B LDS load per lane) + b_hh
    CFrag c;
    {
      O8 xi;
      xi.q = *reinterpret_cast<const uint4*>(sxin[cur] + wave * 256 + lane * 8);
#pragma unroll
      for (int r = 0; r < 8; ++r) c.f[r] = (float)xi.h[r] + bias;
    }

    // preload all 8 A fragments of h (16-load clause, latency hidden
    // under the dependent WMMA chain), then accumulate h @ W_hh^T
    HFrag af[8];
#pragma unroll
    for (int k = 0; k < 8; ++k)
      af[k].v = load_a_frag(sh[cur], HID, lane, k * 32);
#pragma unroll
    for (int k = 0; k < 8; ++k)
      c.v = __builtin_amdgcn_wmma_f32_16x16x32_f16(false, af[k].v, false,
                                                   bf[k].v, (short)0, c.v,
                                                   false, false);

    // h_next = tanh(c), write into back buffer (row-major for A-frag reads)
#pragma unroll
    for (int r = 0; r < 8; ++r) {
      const int m = r + 8 * hs;
      sh[nxt][m * HID + nbase + n] = (_Float16)tanhf(c.f[r]);
    }

    // deposit prefetched xin tile for t+1 into back buffer
    reinterpret_cast<uint4*>(sxin[nxt])[tid] = xreg;

    __syncthreads();
    cur = nxt;
  }

  // write final hidden state tile (16x256 f16) coalesced
  reinterpret_cast<uint4*>(hT + (size_t)mbase * HID)[tid] =
      reinterpret_cast<const uint4*>(sh[cur])[tid];
}

// ---------------------------------------------------------------------------
// Kernel C: out = sigmoid(hT @ W_lin^T + b_lin), fp32 output.
// Grid = 64 M-tiles x 4 N-blocks; each wave one 16x16 tile; B frags hoisted.
// ---------------------------------------------------------------------------
__global__ __launch_bounds__(512) void rnn_head_kernel(
    const _Float16* __restrict__ hT, const _Float16* __restrict__ wlin,
    const float* __restrict__ b_lin, float* __restrict__ out) {
  __shared__ _Float16 sA[16 * HID];

  const int tid   = threadIdx.x;
  const int mt    = blockIdx.x >> 2;
  const int nb    = blockIdx.x & 3;
  const int mbase = mt * 16;
  const int lane  = tid & 31;
  const int wave  = tid >> 5;
  const int n     = lane & 15;
  const int hs    = lane >> 4;
  const int nbase = nb * 256 + wave * 16;

  HFrag bf[8];
#pragma unroll
  for (int k = 0; k < 8; ++k)
    bf[k].v = load_b_frag(wlin + (size_t)nbase * HID, HID, lane, k * 32);

  reinterpret_cast<uint4*>(sA)[tid] =
      reinterpret_cast<const uint4*>(hT + (size_t)mbase * HID)[tid];
  __syncthreads();

  HFrag af[8];
#pragma unroll
  for (int k = 0; k < 8; ++k) af[k].v = load_a_frag(sA, HID, lane, k * 32);

  CFrag c;
  const float bias = b_lin[nbase + n];
#pragma unroll
  for (int r = 0; r < 8; ++r) c.f[r] = bias;
#pragma unroll
  for (int k = 0; k < 8; ++k)
    c.v = __builtin_amdgcn_wmma_f32_16x16x32_f16(false, af[k].v, false,
                                                 bf[k].v, (short)0, c.v,
                                                 false, false);

#pragma unroll
  for (int r = 0; r < 8; ++r) {
    const int m = r + 8 * hs;
    const float v = c.f[r];
    out[(size_t)(mbase + m) * OUT_DIM + nbase + n] = 1.0f / (1.0f + expf(-v));
  }
}

// ---------------------------------------------------------------------------
// Host launcher. Workspace layout (bytes):
//   [0)        W_ih  f16 : 131072
//   [131072)   W_hh  f16 : 131072
//   [262144)   W_lin f16 : 524288
//   [786432)   hT    f16 : 524288
//   [1310720)  xin   f16 : 268435456   (total ~257.1 MB)
// ---------------------------------------------------------------------------
extern "C" void kernel_launch(void* const* d_in, const int* in_sizes, int n_in,
                              void* d_out, int out_size, void* d_ws,
                              size_t ws_size, hipStream_t stream) {
  (void)in_sizes; (void)n_in; (void)out_size; (void)ws_size;
  const float* x     = (const float*)d_in[0];
  const float* W_ih  = (const float*)d_in[1];
  const float* W_hh  = (const float*)d_in[2];
  const float* b_ih  = (const float*)d_in[3];
  const float* b_hh  = (const float*)d_in[4];
  const float* W_lin = (const float*)d_in[5];
  const float* b_lin = (const float*)d_in[6];
  float* out = (float*)d_out;

  char* ws = (char*)d_ws;
  _Float16* wih_h  = (_Float16*)(ws + 0);
  _Float16* whh_h  = (_Float16*)(ws + 131072);
  _Float16* wlin_h = (_Float16*)(ws + 262144);
  _Float16* hT     = (_Float16*)(ws + 786432);
  _Float16* xin    = (_Float16*)(ws + 1310720);

  // weight conversions (tiny)
  cvt_f32_f16_kernel<<<(HID * IN_DIM + 255) / 256, 256, 0, stream>>>(
      W_ih, wih_h, HID * IN_DIM);
  cvt_f32_f16_kernel<<<(HID * HID + 255) / 256, 256, 0, stream>>>(
      W_hh, whh_h, HID * HID);
  cvt_f32_f16_kernel<<<(OUT_DIM * HID + 255) / 256, 256, 0, stream>>>(
      W_lin, wlin_h, OUT_DIM * HID);

  // input projection GEMM: (T*B)/(16*PROJ_MT) = 8192 blocks
  rnn_proj_kernel<<<(T_STEPS * BATCH) / (16 * PROJ_MT), 512, 0, stream>>>(
      x, wih_h, b_ih, xin);

  // recurrence: 64 blocks x 512 threads, 32 KB static LDS
  rnn_recur_kernel<<<BATCH / 16, 512, 0, stream>>>(xin, whh_h, b_hh, hT);

  // output head: 64 M-tiles x 4 N-blocks
  rnn_head_kernel<<<(BATCH / 16) * (OUT_DIM / 256), 512, 0, stream>>>(
      hT, wlin_h, b_lin, out);
}
